// CoHGCLModel_87668872446041
// MI455X (gfx1250) — matrix operand order repaired and checked by reading, not compiled, not run on hardware
//
#include <hip/hip_runtime.h>

// ---------------------------------------------------------------------------
// CDNA5 (gfx1250) implementation of the CoHGCL forward pass.
// Dense matmuls: v_wmma_f32_16x16x32_bf16 (fp32 -> bf16 inputs, f32 accum).
// A tiles staged via the Tensor Data Mover (tensor_load_to_lds + TENSORcnt).
// InfoNCE is a fused flash-style WMMA logsumexp kernel.
// ---------------------------------------------------------------------------

typedef __attribute__((ext_vector_type(16))) __bf16 v16bf;
typedef __attribute__((ext_vector_type(8)))  float  v8f;
typedef unsigned int u32x4 __attribute__((ext_vector_type(4)));
typedef int          i32x4 __attribute__((ext_vector_type(4)));
typedef int          i32x8 __attribute__((ext_vector_type(8)));

union Frag16 {          // one WMMA bf16 operand fragment per lane (32 bytes)
    v16bf v;
    uint4 u[2];
};

__device__ inline v16bf cvt16(float4 a, float4 b, float4 c, float4 d) {
    v16bf v;
    v[0]  = (__bf16)a.x; v[1]  = (__bf16)a.y; v[2]  = (__bf16)a.z; v[3]  = (__bf16)a.w;
    v[4]  = (__bf16)b.x; v[5]  = (__bf16)b.y; v[6]  = (__bf16)b.z; v[7]  = (__bf16)b.w;
    v[8]  = (__bf16)c.x; v[9]  = (__bf16)c.y; v[10] = (__bf16)c.z; v[11] = (__bf16)c.w;
    v[12] = (__bf16)d.x; v[13] = (__bf16)d.y; v[14] = (__bf16)d.z; v[15] = (__bf16)d.w;
    return v;
}

__device__ inline v16bf cvt8x2(const float* __restrict__ a,
                               const float* __restrict__ b) {
    v16bf v;
#pragma unroll
    for (int i = 0; i < 8; ++i) v[i] = (__bf16)a[i];
#pragma unroll
    for (int i = 0; i < 8; ++i) v[8 + i] = (__bf16)b[i];
    return v;
}

// ---------------------------------------------------------------------------
// Tensor Data Mover: 2D tile load Global->LDS (D# per CDNA5 ISA ch.8).
//   tile_x elements per row, tile_y rows; tensor_x/tensor_y = valid extent
//   from the tile origin (OOB reads return zero -> zero padding for free);
//   stride = elements between rows; dsz_log2: 0=1B,1=2B,2=4B.
// LDS receives the tile linearly: [y][x], row stride = tile_x.
// ---------------------------------------------------------------------------
#if __has_builtin(__builtin_amdgcn_tensor_load_to_lds)
#define HAVE_TDM 1
__device__ inline void tdm_load_2d(void* lds_ptr, const void* gptr,
                                   int tile_x, int tile_y,
                                   long tensor_x, long tensor_y,
                                   long stride_elems, int dsz_log2) {
    unsigned lds_addr = (unsigned)(size_t)lds_ptr;          // low 32 bits = LDS offset
    unsigned long long ga = (unsigned long long)(size_t)gptr;
    unsigned tx = (unsigned)tensor_x, ty = (unsigned)tensor_y;
    unsigned long long st = (unsigned long long)stride_elems;

    u32x4 g0;
    g0[0] = 1u;                                             // count=1, is_restore=0
    g0[1] = lds_addr;                                       // lds_addr [63:32]
    g0[2] = (unsigned)(ga & 0xFFFFFFFFu);                   // global_addr [95:64]
    g0[3] = (unsigned)((ga >> 32) & 0x1FFFFFFu) | (2u << 30); // addr[56:32] | type=2

    i32x8 g1;
    g1[0] = (dsz_log2 & 3) << 16;                           // wg_mask=0 | data_size
    g1[1] = (int)(tx << 16);                                // tensor_dim0[15:0]
    g1[2] = (int)((tx >> 16) & 0xFFFFu) | (int)(ty << 16);  // dim0[31:16] | dim1[15:0]
    g1[3] = (int)((ty >> 16) & 0xFFFFu) | ((tile_x & 0xFFFF) << 16); // dim1[31:16]|tile0
    g1[4] = (tile_y & 0xFFFF);                              // tile_dim1 (tile_dim2=0)
    g1[5] = (int)(st & 0xFFFFFFFFull);                      // dim0_stride[31:0]
    g1[6] = (int)((st >> 32) & 0xFFFFull);                  // dim0_stride[47:32]
    g1[7] = 0;

    i32x4 z4 = {0, 0, 0, 0};
#if __clang_major__ >= 23
    i32x8 z8 = {0, 0, 0, 0, 0, 0, 0, 0};
    __builtin_amdgcn_tensor_load_to_lds(g0, g1, z4, z4, z8, 0);
#else
    __builtin_amdgcn_tensor_load_to_lds(g0, g1, z4, z4, 0);
#endif
}
#else
#define HAVE_TDM 0
#endif

// ---------------------------------------------------------------------------
// WMMA GEMM:  C[M x N] = op(A) * op(B),  optional ELU epilogue.
//   TA=0: A [M x K] row-major;  TA=1: A stored [K x M]
//   TB=0: B [K x N];            TB=1: B stored [N x K]
// 256 threads (8 waves). Block tile 128x64, K-step 32.
// A tile (fp32) is TDM-staged; B tile converted to bf16 n-major by the block.
// Wave w computes rows [w*16, w*16+16), all 64 block columns (4 WMMA/K-step).
// ---------------------------------------------------------------------------
#define TBM 128
#define TBN 64
#define TBK 32

template <int TA, int TB>
__global__ __launch_bounds__(256)
void wmma_gemm(const float* __restrict__ A, const float* __restrict__ B,
               float* __restrict__ C, int M, int N, int K, int epi) {
    __shared__ __align__(16) float  Afs[TBM * TBK];  // TA=0: [m][k]; TA=1: [k][m]
    __shared__ __align__(16) __bf16 Bs[TBN * TBK];   // n-major [n][k]

    const int tid  = threadIdx.x;
    const int lane = tid & 31;
    const int wave = tid >> 5;
    const int bm   = blockIdx.y * TBM;
    const int bn   = blockIdx.x * TBN;

    v8f acc[4] = {};

    const int kTiles = (K + TBK - 1) / TBK;
    for (int kt = 0; kt < kTiles; ++kt) {
        const int k0 = kt * TBK;

        // ---- issue async TDM load of the A tile (fp32, OOB zero-filled) ----
#if HAVE_TDM
        if (tid == 0) {
            if (TA == 0)   // tile = rows [bm,bm+128) x cols [k0,k0+32) of A[MxK]
                tdm_load_2d(Afs, A + (size_t)bm * K + k0, TBK, TBM,
                            (long)K - k0, (long)M - bm, K, 2);
            else           // tile = rows [k0,k0+32) x cols [bm,bm+128) of A[KxM]
                tdm_load_2d(Afs, A + (size_t)k0 * M + bm, TBM, TBK,
                            (long)M - bm, (long)K - k0, M, 2);
        }
#else
        for (int i = tid; i < TBM * TBK; i += 256) {
            int r = i / TBK, c = i % TBK;
            int gr = bm + r, gc = k0 + c;
            float v = 0.f;
            if (gr < M && gc < K)
                v = TA ? A[(size_t)gc * M + gr] : A[(size_t)gr * K + gc];
            if (TA == 0) Afs[r * TBK + c] = v;     // [m][k]
            else         Afs[c * TBM + r] = v;     // wait-free remap below unused
        }
#endif

        // ---- stage B tile into bf16 n-major while the TDM runs ----
        {
            // 2048 elements, 8 per thread: n = tid/4, k-chunk = (tid%4)*8
            int n  = tid >> 2;
            int c  = (tid & 3) * 8;
            int gn = bn + n;
            int gc = k0 + c;
            float va[8];
            if (TB == 1) {                 // B[N x K]: contiguous in K
                if (gn < N && gc + 7 < K) {
                    float4 x0 = *(const float4*)(B + (size_t)gn * K + gc);
                    float4 x1 = *(const float4*)(B + (size_t)gn * K + gc + 4);
                    va[0] = x0.x; va[1] = x0.y; va[2] = x0.z; va[3] = x0.w;
                    va[4] = x1.x; va[5] = x1.y; va[6] = x1.z; va[7] = x1.w;
                } else {
#pragma unroll
                    for (int j = 0; j < 8; ++j)
                        va[j] = (gn < N && gc + j < K) ? B[(size_t)gn * K + gc + j] : 0.f;
                }
            } else {                       // B[K x N]: gather down a column
#pragma unroll
                for (int j = 0; j < 8; ++j)
                    va[j] = (gn < N && gc + j < K) ? B[(size_t)(gc + j) * N + gn] : 0.f;
            }
            union { __bf16 h[8]; uint4 u; } pk;
#pragma unroll
            for (int j = 0; j < 8; ++j) pk.h[j] = (__bf16)va[j];
            *(uint4*)(Bs + n * TBK + c) = pk.u;
        }

        // prefetch next B K-chunk into GL2
        if (kt + 1 < kTiles) {
            int gn = bn + (tid >> 2), gc = k0 + TBK + (tid & 3) * 8;
            if (gn < N && gc < K)
                __builtin_prefetch(TB ? (const void*)&B[(size_t)gn * K + gc]
                                      : (const void*)&B[(size_t)gc * N + gn], 0, 0);
        }

#if HAVE_TDM
        if (tid == 0) __builtin_amdgcn_s_wait_tensorcnt(0);
#endif
        __syncthreads();

        // ---- A fragment (fp32 LDS -> bf16), reused for 4 column tiles ----
        Frag16 af;
        {
            int m  = wave * 16 + (lane & 15);
            int b0 = (lane < 16) ? 0 : 8;
            if (TA == 0) {
                const float* row = Afs + m * TBK;
                af.v = cvt16(*(const float4*)(row + b0),
                             *(const float4*)(row + b0 + 4),
                             *(const float4*)(row + b0 + 16),
                             *(const float4*)(row + b0 + 20));
            } else {       // LDS is [k][m], k-stride TBM
#pragma unroll
                for (int e = 0; e < 8; ++e)
                    af.v[e] = (__bf16)Afs[(b0 + e) * TBM + m];
#pragma unroll
                for (int e = 0; e < 8; ++e)
                    af.v[8 + e] = (__bf16)Afs[(b0 + 16 + e) * TBM + m];
            }
        }
        const int kb = (lane < 16) ? 0 : 8;
#pragma unroll
        for (int ct = 0; ct < 4; ++ct) {
            Frag16 bfr;
            const __bf16* rowb = Bs + (ct * 16 + (lane & 15)) * TBK + kb;
            bfr.u[0] = *(const uint4*)(rowb);
            bfr.u[1] = *(const uint4*)(rowb + 16);
            acc[ct] = __builtin_amdgcn_wmma_f32_16x16x32_bf16(
                false, af.v, false, bfr.v, (short)0, acc[ct], false, false);
        }
        __syncthreads();
    }

    // ---- writeback (C/D layout: VGPR r -> M=r(+8), N=lane&15) ----
    const int mOff  = bm + wave * 16 + ((lane < 16) ? 0 : 8);
    const int nCol0 = bn + (lane & 15);
#pragma unroll
    for (int ct = 0; ct < 4; ++ct) {
        int col = nCol0 + ct * 16;
        if (col >= N) continue;
#pragma unroll
        for (int r = 0; r < 8; ++r) {
            int row = mOff + r;
            if (row < M) {
                float v = acc[ct][r];
                if (epi == 1) v = (v > 0.f) ? v : (__expf(v) - 1.f);
                C[(size_t)row * N + col] = v;
            }
        }
    }
}

// ---------------------------------------------------------------------------
// Fused InfoNCE: per 16-row tile of Z1, stream column tiles of Z2, WMMA the
// 16x16 sim tile, online row-wise logsumexp, capture the diagonal, atomically
// accumulate mean(-log p[i,i]).  Z1/Z2 row-normalized [n x 128], n % 16 == 0.
// ---------------------------------------------------------------------------
__global__ __launch_bounds__(32)
void infonce_rows(const float* __restrict__ Z1, const float* __restrict__ Z2,
                  int n, float invTemp, float invN, float* __restrict__ acc_out) {
    const int tile = blockIdx.x;
    const int lane = threadIdx.x;

    Frag16 af[4];
    {
        int m  = tile * 16 + (lane & 15);
        int b0 = (lane < 16) ? 0 : 8;
        const float* arow = Z1 + (size_t)m * 128;
#pragma unroll
        for (int kk = 0; kk < 4; ++kk)
            af[kk].v = cvt8x2(arow + kk * 32 + b0, arow + kk * 32 + b0 + 16);
    }

    float rmax[8], rsum[8], rdiag[8];
#pragma unroll
    for (int r = 0; r < 8; ++r) { rmax[r] = -3.0e38f; rsum[r] = 0.f; rdiag[r] = 0.f; }

    const int kb = (lane < 16) ? 0 : 8;
    const int cTiles = n / 16;
    for (int ct = 0; ct < cTiles; ++ct) {
        v8f acc = {};
        const float* brow = Z2 + (size_t)(ct * 16 + (lane & 15)) * 128;
#pragma unroll
        for (int kk = 0; kk < 4; ++kk) {
            Frag16 bfr;
            bfr.v = cvt8x2(brow + kk * 32 + kb, brow + kk * 32 + kb + 16);
            acc = __builtin_amdgcn_wmma_f32_16x16x32_bf16(
                false, af[kk].v, false, bfr.v, (short)0, acc, false, false);
        }
        const bool diagTile = (ct == tile);
#pragma unroll
        for (int r = 0; r < 8; ++r) {
            float x = acc[r] * invTemp;
            if (diagTile) {
                int myM = r + ((lane < 16) ? 0 : 8);
                if ((lane & 15) == myM) rdiag[r] = x;
            }
            float tmax = x;
            tmax = fmaxf(tmax, __shfl_xor(tmax, 1, 32));
            tmax = fmaxf(tmax, __shfl_xor(tmax, 2, 32));
            tmax = fmaxf(tmax, __shfl_xor(tmax, 4, 32));
            tmax = fmaxf(tmax, __shfl_xor(tmax, 8, 32));
            float mnew = fmaxf(rmax[r], tmax);
            float ex = __expf(x - mnew);
            float ts = ex;
            ts += __shfl_xor(ts, 1, 32);
            ts += __shfl_xor(ts, 2, 32);
            ts += __shfl_xor(ts, 4, 32);
            ts += __shfl_xor(ts, 8, 32);
            rsum[r] = rsum[r] * __expf(rmax[r] - mnew) + ts;
            rmax[r] = mnew;
        }
    }
#pragma unroll
    for (int r = 0; r < 8; ++r) {
        int myM = r + ((lane < 16) ? 0 : 8);
        if ((lane & 15) == myM) {
            float loss = logf(rsum[r]) + rmax[r] - rdiag[r];  // -log p[i,i]
            atomicAdd(acc_out, loss * invN);
        }
    }
}

// ---------------------------------------------------------------------------
// Small helper kernels
// ---------------------------------------------------------------------------
__global__ void vec_zero(float* p, int n) {
    int i = blockIdx.x * blockDim.x + threadIdx.x;
    if (i < n) p[i] = 0.f;
}
__global__ void vec_copy(const float* __restrict__ a, float* __restrict__ b, int n) {
    int i = blockIdx.x * blockDim.x + threadIdx.x;
    if (i < n) b[i] = a[i];
}
__global__ void vec_add(const float* __restrict__ a, const float* __restrict__ b,
                        float* __restrict__ c, int n) {
    int i = blockIdx.x * blockDim.x + threadIdx.x;
    if (i < n) c[i] = a[i] + b[i];
}
__global__ void vec_elu(float* p, int n) {
    int i = blockIdx.x * blockDim.x + threadIdx.x;
    if (i < n) { float v = p[i]; p[i] = (v > 0.f) ? v : (__expf(v) - 1.f); }
}

// one wave per row: out[row] = dot(X[row, 0:128], a[0:128])
__global__ __launch_bounds__(256)
void rowdot128(const float* __restrict__ X, const float* __restrict__ a,
               float* __restrict__ out, int n) {
    int row  = blockIdx.x * (blockDim.x >> 5) + (threadIdx.x >> 5);
    int lane = threadIdx.x & 31;
    if (row >= n) return;
    float4 x  = ((const float4*)(X + (size_t)row * 128))[lane];
    float4 aa = ((const float4*)a)[lane];
    float s = x.x * aa.x + x.y * aa.y + x.z * aa.z + x.w * aa.w;
    for (int m = 16; m > 0; m >>= 1) s += __shfl_xor(s, m, 32);
    if (lane == 0) out[row] = s;
}

// one wave per row: Y[row] = X[row] / (||X[row]|| + 1e-12)
__global__ __launch_bounds__(256)
void rownorm128(const float* __restrict__ X, float* __restrict__ Y, int n) {
    int row  = blockIdx.x * (blockDim.x >> 5) + (threadIdx.x >> 5);
    int lane = threadIdx.x & 31;
    if (row >= n) return;
    float4 x = ((const float4*)(X + (size_t)row * 128))[lane];
    float s = x.x * x.x + x.y * x.y + x.z * x.z + x.w * x.w;
    for (int m = 16; m > 0; m >>= 1) s += __shfl_xor(s, m, 32);
    float inv = 1.f / (sqrtf(s) + 1e-12f);
    float4 y = { x.x * inv, x.y * inv, x.z * inv, x.w * inv };
    ((float4*)(Y + (size_t)row * 128))[lane] = y;
}

// edge pass 1: exp(clip(leakyrelu(s_i[s]+s_j[t]))), accumulate denom[s]
__global__ void edge_softmax_num(const float* __restrict__ si, const float* __restrict__ sj,
                                 const int* __restrict__ sidx, const int* __restrict__ tidx,
                                 float* __restrict__ exp_e, float* __restrict__ denom, int E) {
    int e = blockIdx.x * blockDim.x + threadIdx.x;
    if (e >= E) return;
    int s = sidx[e], t = tidx[e];
    float v = si[s] + sj[t];
    v = (v > 0.f) ? v : 0.2f * v;
    v = fminf(30.f, fmaxf(-30.f, v));
    float ex = __expf(v);
    exp_e[e] = ex;
    atomicAdd(&denom[s], ex);
}

// edge pass 2: out[s] += alpha * h_j[t]  (one wave per edge, float4 lanes)
__global__ __launch_bounds__(256)
void edge_aggregate(const float* __restrict__ hj, const float* __restrict__ exp_e,
                    const float* __restrict__ denom,
                    const int* __restrict__ sidx, const int* __restrict__ tidx,
                    float* __restrict__ out, int E) {
    int wid  = (int)((blockIdx.x * (size_t)blockDim.x + threadIdx.x) >> 5);
    int lane = threadIdx.x & 31;
    if (wid >= E) return;
    int s = sidx[wid], t = tidx[wid];
    float alpha = exp_e[wid] / (denom[s] + 1e-8f);
    float4 v = ((const float4*)(hj + (size_t)t * 128))[lane];
    float* o = out + (size_t)s * 128 + lane * 4;
    atomicAdd(o + 0, v.x * alpha);
    atomicAdd(o + 1, v.y * alpha);
    atomicAdd(o + 2, v.z * alpha);
    atomicAdd(o + 3, v.w * alpha);
}

// fused GMF + MLP + gated fusion; one 128-thread block per (t,d) pair
__global__ __launch_bounds__(128)
void fuse_predict(const float* __restrict__ final_t, const float* __restrict__ final_d,
                  const int* __restrict__ t_ids, const int* __restrict__ d_ids,
                  const float* __restrict__ W1, const float* __restrict__ b1,
                  const float* __restrict__ W2, const float* __restrict__ b2,
                  const float* __restrict__ gW, const float* __restrict__ gb,
                  float* __restrict__ fusion, float* __restrict__ gmf,
                  float* __restrict__ mlp, int B) {
    __shared__ float s_in[256];
    __shared__ float rg[4], rm[4], rp[4];
    int p = blockIdx.x;
    if (p >= B) return;
    int tid = threadIdx.x;
    int ti = t_ids[p], di = d_ids[p];
    float tv = final_t[(size_t)ti * 128 + tid];
    float dv = final_d[(size_t)di * 128 + tid];
    s_in[tid] = tv;
    s_in[128 + tid] = dv;
    float g = tv * dv;
    __syncthreads();

    float acc = b1[tid];
    const float* w = W1 + (size_t)tid * 256;
#pragma unroll 8
    for (int j = 0; j < 256; ++j) acc += w[j] * s_in[j];
    float h1 = fmaxf(acc, 0.f);

    float sg = g, sm = h1 * W2[tid], sp = g * gW[tid];
    for (int m = 16; m > 0; m >>= 1) {
        sg += __shfl_xor(sg, m, 32);
        sm += __shfl_xor(sm, m, 32);
        sp += __shfl_xor(sp, m, 32);
    }
    int wv = tid >> 5, lane = tid & 31;
    if (lane == 0) { rg[wv] = sg; rm[wv] = sm; rp[wv] = sp; }
    __syncthreads();
    if (tid == 0) {
        float G  = rg[0] + rg[1] + rg[2] + rg[3];
        float Mp = rm[0] + rm[1] + rm[2] + rm[3] + b2[0];
        float Gt = rp[0] + rp[1] + rp[2] + rp[3] + gb[0];
        float gate = 1.f / (1.f + __expf(-Gt));
        gmf[p] = G;
        mlp[p] = Mp;
        fusion[p] = gate * G + (1.f - gate) * Mp;
    }
}

// ---------------------------------------------------------------------------
// Host-side orchestration
// ---------------------------------------------------------------------------
static inline void launch_gemm(const float* A, const float* B, float* C,
                               int M, int N, int K, int tA, int tB, int epi,
                               hipStream_t s) {
    dim3 grid((N + TBN - 1) / TBN, (M + TBM - 1) / TBM);
    if (!tA && !tB)      wmma_gemm<0, 0><<<grid, 256, 0, s>>>(A, B, C, M, N, K, epi);
    else if (!tA && tB)  wmma_gemm<0, 1><<<grid, 256, 0, s>>>(A, B, C, M, N, K, epi);
    else                 wmma_gemm<1, 0><<<grid, 256, 0, s>>>(A, B, C, M, N, K, epi);
}
static inline void launch_zero(float* p, int n, hipStream_t s) {
    vec_zero<<<(n + 255) / 256, 256, 0, s>>>(p, n);
}

struct GatWs {
    float *h_i, *h_j, *s_i, *s_j, *exp_e, *denom;
};

static void run_gat(const float* src, const float* tgt,
                    const int* sidx, const int* tidx,
                    const float* Wsrc, const float* Wtgt, const float* avec,
                    int n_src, int n_tgt, int E, float* out,
                    const GatWs& w, hipStream_t s) {
    launch_gemm(src, Wsrc, w.h_i, n_src, 128, 128, 0, 1, 0, s);   // h_i = src @ Wsrc^T
    launch_gemm(tgt, Wtgt, w.h_j, n_tgt, 128, 128, 0, 1, 0, s);   // h_j = tgt @ Wtgt^T
    rowdot128<<<(n_src + 7) / 8, 256, 0, s>>>(w.h_i, avec, w.s_i, n_src);
    rowdot128<<<(n_tgt + 7) / 8, 256, 0, s>>>(w.h_j, avec + 128, w.s_j, n_tgt);
    launch_zero(w.denom, n_src, s);
    edge_softmax_num<<<(E + 255) / 256, 256, 0, s>>>(w.s_i, w.s_j, sidx, tidx,
                                                     w.exp_e, w.denom, E);
    launch_zero(out, n_src * 128, s);
    edge_aggregate<<<(E * 32 + 255) / 256, 256, 0, s>>>(w.h_j, w.exp_e, w.denom,
                                                        sidx, tidx, out, E);
    vec_elu<<<(n_src * 128 + 255) / 256, 256, 0, s>>>(out, n_src * 128);
}

static void run_hg(const float* x, const float* hyper, int n, int H,
                   float* Sbuf, float* Tm, float* out, hipStream_t s) {
    launch_gemm(x, hyper, Sbuf, n, H, 128, 0, 0, 0, s);      // S = x @ hyper
    launch_gemm(Sbuf, x, Tm, H, 128, n, 1, 0, 0, s);         // T = S^T @ x
    launch_gemm(Sbuf, Tm, out, n, 128, H, 0, 0, 1, s);       // out = elu(S @ T)
}

extern "C" void kernel_launch(void* const* d_in, const int* in_sizes, int n_in,
                              void* d_out, int out_size, void* d_ws, size_t ws_size,
                              hipStream_t stream) {
    (void)n_in; (void)out_size; (void)ws_size;
    const int D = 128;
    const float* trna    = (const float*)d_in[0];
    const float* disease = (const float*)d_in[1];
    const float* thyper  = (const float*)d_in[2];
    const float* dhyper  = (const float*)d_in[3];
    const float* WtS     = (const float*)d_in[4];   // [3][128][128]
    const float* WtT     = (const float*)d_in[5];
    const float* aT      = (const float*)d_in[6];   // [3][256]
    const float* WdS     = (const float*)d_in[7];
    const float* WdT     = (const float*)d_in[8];
    const float* aD      = (const float*)d_in[9];
    const float* W1      = (const float*)d_in[10];
    const float* b1      = (const float*)d_in[11];
    const float* W2      = (const float*)d_in[12];
    const float* b2      = (const float*)d_in[13];
    const float* gW      = (const float*)d_in[14];
    const float* gb      = (const float*)d_in[15];
    const int*   src_id  = (const int*)d_in[16];
    const int*   tgt_id  = (const int*)d_in[17];
    const int*   t_ids   = (const int*)d_in[18];
    const int*   d_ids   = (const int*)d_in[19];

    const int TRNA = in_sizes[0] / D;      // 8000
    const int DIS  = in_sizes[1] / D;      // 6000
    const int H    = in_sizes[2] / D;      // 256
    const int E    = in_sizes[16];         // 320000
    const int B    = in_sizes[18];         // 16384

    // ---- workspace carve ----
    float* ws = (float*)d_ws;
    size_t off = 0;
    auto alloc = [&](size_t n) { float* p = ws + off; off += n; return p; };
    float* h_i     = alloc((size_t)TRNA * D);
    float* h_j     = alloc((size_t)TRNA * D);
    float* s_i     = alloc(TRNA);
    float* s_j     = alloc(TRNA);
    float* exp_e   = alloc(E);
    float* denom   = alloc(TRNA);
    float* sg_t    = alloc((size_t)TRNA * D);
    float* sg_d    = alloc((size_t)DIS * D);
    float* hg_t    = alloc((size_t)TRNA * D);
    float* hg_d    = alloc((size_t)DIS * D);
    float* S_t     = alloc((size_t)TRNA * H);
    float* S_d     = alloc((size_t)DIS * H);
    float* Tm      = alloc((size_t)H * D);
    float* z1n     = alloc((size_t)TRNA * D);
    float* z2n     = alloc((size_t)TRNA * D);
    float* final_t = alloc((size_t)TRNA * D);
    float* final_d = alloc((size_t)DIS * D);
    float* tmp_t   = alloc((size_t)TRNA * D);
    float* tmp_d   = alloc((size_t)DIS * D);

    GatWs gw{h_i, h_j, s_i, s_j, exp_e, denom};

    // ---- output carve (fusion, sg_t, sg_d, gmf, mlp, contrast) ----
    float* out_fusion   = (float*)d_out;
    float* out_sgt      = out_fusion + B;
    float* out_sgd      = out_sgt + (size_t)TRNA * D;
    float* out_gmf      = out_sgd + (size_t)DIS * D;
    float* out_mlp      = out_gmf + B;
    float* out_contrast = out_mlp + B;

    // ---- GAT layer 1 ----
    run_gat(trna, disease, src_id, tgt_id, WtS, WtT, aT, TRNA, DIS, E, sg_t, gw, stream);
    run_gat(disease, sg_t, tgt_id, src_id, WdS, WdT, aD, DIS, TRNA, E, sg_d, gw, stream);

    // ---- hypergraph passes 1-2 ----
    run_hg(sg_t, thyper, TRNA, H, S_t, Tm, hg_t, stream);
    run_hg(sg_d, dhyper, DIS,  H, S_d, Tm, hg_d, stream);
    run_hg(hg_t, thyper, TRNA, H, S_t, Tm, hg_t, stream);
    run_hg(hg_d, dhyper, DIS,  H, S_d, Tm, hg_d, stream);

    // ---- GAT layers 2-3 ----
    run_gat(hg_t, hg_d, src_id, tgt_id, WtS + 1 * D * D, WtT + 1 * D * D, aT + 1 * 2 * D,
            TRNA, DIS, E, tmp_t, gw, stream);
    run_gat(hg_d, hg_t, tgt_id, src_id, WdS + 1 * D * D, WdT + 1 * D * D, aD + 1 * 2 * D,
            DIS, TRNA, E, tmp_d, gw, stream);
    run_gat(tmp_t, tmp_d, src_id, tgt_id, WtS + 2 * D * D, WtT + 2 * D * D, aT + 2 * 2 * D,
            TRNA, DIS, E, sg_t, gw, stream);
    run_gat(tmp_d, sg_t, tgt_id, src_id, WdS + 2 * D * D, WdT + 2 * D * D, aD + 2 * 2 * D,
            DIS, TRNA, E, sg_d, gw, stream);

    // ---- hypergraph pass 3 ----
    run_hg(hg_t, thyper, TRNA, H, S_t, Tm, hg_t, stream);
    run_hg(hg_d, dhyper, DIS,  H, S_d, Tm, hg_d, stream);

    // ---- final embeddings ----
    vec_add<<<((TRNA * D) + 255) / 256, 256, 0, stream>>>(sg_t, hg_t, final_t, TRNA * D);
    vec_add<<<((DIS * D) + 255) / 256, 256, 0, stream>>>(sg_d, hg_d, final_d, DIS * D);

    // ---- contrastive loss (fused WMMA logsumexp) ----
    launch_zero(out_contrast, 1, stream);
    rownorm128<<<(TRNA + 7) / 8, 256, 0, stream>>>(sg_t, z1n, TRNA);
    rownorm128<<<(TRNA + 7) / 8, 256, 0, stream>>>(hg_t, z2n, TRNA);
    infonce_rows<<<TRNA / 16, 32, 0, stream>>>(z1n, z2n, TRNA, 2.0f, 1.0f / TRNA, out_contrast);
    rownorm128<<<(DIS + 7) / 8, 256, 0, stream>>>(sg_d, z1n, DIS);
    rownorm128<<<(DIS + 7) / 8, 256, 0, stream>>>(hg_d, z2n, DIS);
    infonce_rows<<<DIS / 16, 32, 0, stream>>>(z1n, z2n, DIS, 2.0f, 1.0f / DIS, out_contrast);

    // ---- sg outputs ----
    vec_copy<<<((TRNA * D) + 255) / 256, 256, 0, stream>>>(sg_t, out_sgt, TRNA * D);
    vec_copy<<<((DIS * D) + 255) / 256, 256, 0, stream>>>(sg_d, out_sgd, DIS * D);

    // ---- fused GMF / MLP / gate predictions ----
    fuse_predict<<<B, 128, 0, stream>>>(final_t, final_d, t_ids, d_ids,
                                        W1, b1, W2, b2, gW, gb,
                                        out_fusion, out_gmf, out_mlp, B);
}
